// BasicBlock_49263274885742
// MI455X (gfx1250) — matrix operand order repaired and checked by reading, not compiled, not run on hardware
//
#include <hip/hip_runtime.h>
#include <hip/hip_bf16.h>

// ---------------------------------------------------------------------------
// CDNA5 (gfx1250) implementation: edge MLP + segment-max + node MLP, all
// GEMM work on v_wmma_f32_16x16x32_f16 (wave32 WMMA, f32 accumulate).
// ---------------------------------------------------------------------------

typedef __attribute__((ext_vector_type(16))) _Float16 v16h;
typedef __attribute__((ext_vector_type(8)))  _Float16 h8;
typedef __attribute__((ext_vector_type(8)))  float    v8f;

#define HID 64
#define K1P 96   // layer-1 K (=67) padded to a multiple of 32

// ---- WMMA fragment helpers (layouts per CDNA5 ISA 7.12.2, wave32) ---------
// A (16x32 f16): lane L holds row M=L%16; element e: K = kb*32 + 8*(L/16) + e
// for e<8, and K = kb*32 + 16 + 8*(L/16) + (e-8) for e>=8.  => two runs of 8
// consecutive halves => two 16-byte LDS loads.
// B (32x16 f16): lane L holds col N=L%16; element e: K as for A above (two runs of 8)
// => two runs of 8 consecutive halves of a column => two 16-byte loads from a
// column-major weight image.
// C/D (16x16 f32): VGPR r: M = r + 8*(L/16), N = L%16.

__device__ __forceinline__ v16h frag_cat(h8 lo, h8 hi) {
  return __builtin_shufflevector(lo, hi, 0,1,2,3,4,5,6,7,8,9,10,11,12,13,14,15);
}

__device__ __forceinline__ v16h lds_afrag(const _Float16* rowbase, int koff) {
  h8 lo = *(const h8*)(rowbase + koff);
  h8 hi = *(const h8*)(rowbase + koff + 16);
  return frag_cat(lo, hi);
}

__device__ __forceinline__ v16h lds_bfrag(const _Float16* wcol, int kpad,
                                          int n, int kb, int hg) {
  const _Float16* p = wcol + n * kpad + kb * 32 + hg * 8;
  h8 lo = *(const h8*)p;
  h8 hi = *(const h8*)(p + 16);
  return frag_cat(lo, hi);
}

#define WMMA_F16(a, b, c) \
  __builtin_amdgcn_wmma_f32_16x16x32_f16(false, (a), false, (b), (short)0, (c), false, false)

// ---------------------------------------------------------------------------
// Kernel 0: convert weights to f16 column-major (B-fragment friendly) images.
// ws layout (halves): W1c [0,6144)  (64 cols x 96 K, rows 67..95 zero)
//                     W2c [6144,10240)   (64 x 64)
//                     W3c [10240,14336)  (64 x 64)
//                     W4c [14336,18432)  (64 x 64)
// ---------------------------------------------------------------------------
__global__ void __launch_bounds__(256) convert_weights_kernel(
    const float* __restrict__ W1, const float* __restrict__ W2,
    const float* __restrict__ W3, const float* __restrict__ W4,
    _Float16* __restrict__ ws) {
  const int i  = blockIdx.x * 256 + threadIdx.x;
  const int n1 = 64 * K1P;   // 6144
  const int n2 = 64 * 64;    // 4096
  if (i < n1) {
    int n = i / K1P, k = i % K1P;
    ws[i] = (_Float16)(k < 67 ? W1[k * 64 + n] : 0.0f);
  } else if (i < n1 + n2) {
    int j = i - n1; int n = j >> 6, k = j & 63;
    ws[i] = (_Float16)W2[k * 64 + n];
  } else if (i < n1 + 2 * n2) {
    int j = i - n1 - n2; int n = j >> 6, k = j & 63;
    ws[i] = (_Float16)W3[k * 64 + n];
  } else if (i < n1 + 3 * n2) {
    int j = i - n1 - 2 * n2; int n = j >> 6, k = j & 63;
    ws[i] = (_Float16)W4[k * 64 + n];
  }
}

// ---------------------------------------------------------------------------
// Kernel 1: per-edge 2-layer MLP via WMMA + scatter-max into agg (= d_out).
// 8 waves / block, each wave owns one 16-edge tile.
// ---------------------------------------------------------------------------
__global__ void __launch_bounds__(256) edge_mlp_kernel(
    const float* __restrict__ last_coors,
    const float* __restrict__ last_features,
    const float* __restrict__ current_coors,
    const int*   __restrict__ ecur,
    const int*   __restrict__ elast,
    const _Float16* __restrict__ wsw,   // W1c @ 0, W2c @ 64*K1P
    const float* __restrict__ b1,
    const float* __restrict__ b2,
    float* __restrict__ agg,
    int ntiles, int E) {
  __shared__ _Float16 sW1[64 * K1P];      // 12 KB
  __shared__ _Float16 sW2[64 * 64];       //  8 KB
  __shared__ _Float16 sX[8][16 * K1P];    // 24 KB  per-wave input staging
  __shared__ _Float16 sH[8][16 * 64];     // 16 KB  per-wave hidden staging

  const int tid = threadIdx.x;
  for (int i = tid; i < 64 * K1P; i += 256) sW1[i] = wsw[i];
  for (int i = tid; i < 64 * 64;  i += 256) sW2[i] = wsw[64 * K1P + i];
  __syncthreads();

  const int wave = tid >> 5;
  const int lane = tid & 31;
  const int r    = lane & 15;   // row within tile / column index for B,D
  const int hg   = lane >> 4;   // half-wave group
  const int tile = blockIdx.x * 8 + wave;
  if (tile >= ntiles) return;   // wave-uniform: EXEC stays all-ones for WMMA

  // ---- stage x = [features(64) | rel(3) | pad->96] for this tile ----------
  {
    int e = tile * 16 + r;
    if (e >= E) e = E - 1;                       // clamp (tail-safe)
    const int li = elast[e];
    const float* f = last_features + (long)li * 64;
    _Float16* xr = &sX[wave][r * K1P];
    if (hg == 0) {
      #pragma unroll
      for (int k = 0; k < 48; ++k) xr[k] = (_Float16)f[k];
    } else {
      const int ci = ecur[e];
      #pragma unroll
      for (int k = 48; k < 64; ++k) xr[k] = (_Float16)f[k];
      #pragma unroll
      for (int j = 0; j < 3; ++j)
        xr[64 + j] = (_Float16)(last_coors[(long)li * 3 + j] -
                                current_coors[(long)ci * 3 + j]);
      #pragma unroll
      for (int k = 67; k < K1P; ++k) xr[k] = (_Float16)0.0f;
    }
  }
  // LDS ops are in-order within a wave; only this wave touches its staging.

  const v8f vzero = {0.f, 0.f, 0.f, 0.f, 0.f, 0.f, 0.f, 0.f};

  // ---- layer 1: [16,96] x [96,64], 3 K-blocks x 4 N-tiles -----------------
  v8f acc[4];
  #pragma unroll
  for (int t = 0; t < 4; ++t) acc[t] = vzero;
  {
    const _Float16* xrow = &sX[wave][r * K1P];
    #pragma unroll
    for (int kb = 0; kb < 3; ++kb) {
      const v16h a = lds_afrag(xrow, kb * 32 + 8 * hg);
      #pragma unroll
      for (int t = 0; t < 4; ++t) {
        const v16h b = lds_bfrag(sW1, K1P, t * 16 + r, kb, hg);
        acc[t] = WMMA_F16(a, b, acc[t]);
      }
    }
  }

  // ---- bias + ReLU, write hidden back through LDS (D->A lane transpose) ---
  #pragma unroll
  for (int t = 0; t < 4; ++t) {
    const int n = t * 16 + r;
    const float bb = b1[n];
    #pragma unroll
    for (int rr = 0; rr < 8; ++rr) {
      const float v = fmaxf(acc[t][rr] + bb, 0.0f);
      sH[wave][(rr + 8 * hg) * 64 + n] = (_Float16)v;
    }
  }

  // ---- layer 2: [16,64] x [64,64] -----------------------------------------
  v8f acc2[4];
  #pragma unroll
  for (int t = 0; t < 4; ++t) acc2[t] = vzero;
  {
    const _Float16* hrow = &sH[wave][r * 64];
    #pragma unroll
    for (int kb = 0; kb < 2; ++kb) {
      const v16h a = lds_afrag(hrow, kb * 32 + 8 * hg);
      #pragma unroll
      for (int t = 0; t < 4; ++t) {
        const v16h b = lds_bfrag(sW2, 64, t * 16 + r, kb, hg);
        acc2[t] = WMMA_F16(a, b, acc2[t]);
      }
    }
  }

  // ---- bias + ReLU + scatter-max (float bits, all values >= 0, base 0) ----
  int ci8[8];
  bool ok8[8];
  #pragma unroll
  for (int rr = 0; rr < 8; ++rr) {
    int e = tile * 16 + rr + 8 * hg;
    ok8[rr] = (e < E);
    if (e >= E) e = E - 1;
    ci8[rr] = ecur[e];
  }
  #pragma unroll
  for (int t = 0; t < 4; ++t) {
    const int n = t * 16 + r;
    const float bb = b2[n];
    #pragma unroll
    for (int rr = 0; rr < 8; ++rr) {
      const float v = fmaxf(acc2[t][rr] + bb, 0.0f);
      if (ok8[rr])
        atomicMax((int*)(agg + (long)ci8[rr] * 64 + n), __float_as_int(v));
    }
  }
}

// ---------------------------------------------------------------------------
// Kernel 2: node 2-layer MLP, in-place on d_out (agg rows are tile-exclusive).
// ---------------------------------------------------------------------------
__global__ void __launch_bounds__(256) out_mlp_kernel(
    const _Float16* __restrict__ wsw,   // W3c @ 0, W4c @ 64*64
    const float* __restrict__ b3,
    const float* __restrict__ b4,
    float* __restrict__ y,              // agg in, result out (in-place)
    int ntiles, int M) {
  __shared__ _Float16 sW3[64 * 64];     // 8 KB
  __shared__ _Float16 sW4[64 * 64];     // 8 KB
  __shared__ _Float16 sH[8][16 * 64];   // 16 KB

  const int tid = threadIdx.x;
  for (int i = tid; i < 64 * 64; i += 256) {
    sW3[i] = wsw[i];
    sW4[i] = wsw[64 * 64 + i];
  }
  __syncthreads();

  const int wave = tid >> 5;
  const int lane = tid & 31;
  const int r    = lane & 15;
  const int hg   = lane >> 4;
  const int tile = blockIdx.x * 8 + wave;
  if (tile >= ntiles) return;

  const v8f vzero = {0.f, 0.f, 0.f, 0.f, 0.f, 0.f, 0.f, 0.f};

  int row = tile * 16 + r;
  if (row >= M) row = M - 1;
  const float* rowp = y + (long)row * 64;

  // ---- layer 3: A fragments straight from global agg rows -----------------
  v8f acc[4];
  #pragma unroll
  for (int t = 0; t < 4; ++t) acc[t] = vzero;
  #pragma unroll
  for (int kb = 0; kb < 2; ++kb) {
    const int k0 = kb * 32 + 8 * hg;
    v16h a;
    #pragma unroll
    for (int i = 0; i < 8; ++i) a[i]     = (_Float16)rowp[k0 + i];
    #pragma unroll
    for (int i = 0; i < 8; ++i) a[i + 8] = (_Float16)rowp[k0 + 16 + i];
    #pragma unroll
    for (int t = 0; t < 4; ++t) {
      const v16h b = lds_bfrag(sW3, 64, t * 16 + r, kb, hg);
      acc[t] = WMMA_F16(a, b, acc[t]);
    }
  }

  // ---- bias + ReLU through LDS --------------------------------------------
  #pragma unroll
  for (int t = 0; t < 4; ++t) {
    const int n = t * 16 + r;
    const float bb = b3[n];
    #pragma unroll
    for (int rr = 0; rr < 8; ++rr) {
      const float v = fmaxf(acc[t][rr] + bb, 0.0f);
      sH[wave][(rr + 8 * hg) * 64 + n] = (_Float16)v;
    }
  }

  // ---- layer 4 -------------------------------------------------------------
  v8f acc2[4];
  #pragma unroll
  for (int t = 0; t < 4; ++t) acc2[t] = vzero;
  {
    const _Float16* hrow = &sH[wave][r * 64];
    #pragma unroll
    for (int kb = 0; kb < 2; ++kb) {
      const v16h a = lds_afrag(hrow, kb * 32 + 8 * hg);
      #pragma unroll
      for (int t = 0; t < 4; ++t) {
        const v16h b = lds_bfrag(sW4, 64, t * 16 + r, kb, hg);
        acc2[t] = WMMA_F16(a, b, acc2[t]);
      }
    }
  }

  // ---- bias + ReLU + store -------------------------------------------------
  #pragma unroll
  for (int t = 0; t < 4; ++t) {
    const int n = t * 16 + r;
    const float bb = b4[n];
    #pragma unroll
    for (int rr = 0; rr < 8; ++rr) {
      const int m = tile * 16 + rr + 8 * hg;
      const float v = fmaxf(acc2[t][rr] + bb, 0.0f);
      if (m < M) y[(long)m * 64 + n] = v;
    }
  }
}

// ---------------------------------------------------------------------------
extern "C" void kernel_launch(void* const* d_in, const int* in_sizes, int n_in,
                              void* d_out, int out_size, void* d_ws, size_t ws_size,
                              hipStream_t stream) {
  const float* last_coors    = (const float*)d_in[0];
  const float* last_features = (const float*)d_in[1];
  const float* current_coors = (const float*)d_in[2];
  const int*   edge          = (const int*)d_in[3];
  const float* W1 = (const float*)d_in[4];  const float* b1 = (const float*)d_in[5];
  const float* W2 = (const float*)d_in[6];  const float* b2 = (const float*)d_in[7];
  const float* W3 = (const float*)d_in[8];  const float* b3 = (const float*)d_in[9];
  const float* W4 = (const float*)d_in[10]; const float* b4 = (const float*)d_in[11];
  float*     out = (float*)d_out;
  _Float16*  wsw = (_Float16*)d_ws;

  const int E = in_sizes[3] / 2;      // edge is [2, E]
  const int M = in_sizes[2] / 3;      // current_coors is [M, 3]

  // d_out doubles as the zero-initialized segment-max accumulator.
  hipMemsetAsync(d_out, 0, (size_t)out_size * sizeof(float), stream);

  const int wcount = 64 * K1P + 3 * 64 * 64;   // halves to convert
  convert_weights_kernel<<<(wcount + 255) / 256, 256, 0, stream>>>(W1, W2, W3, W4, wsw);

  const int etiles = (E + 15) / 16;
  edge_mlp_kernel<<<(etiles + 7) / 8, 256, 0, stream>>>(
      last_coors, last_features, current_coors,
      edge /*cur*/, edge + E /*last*/,
      wsw, b1, b2, out, etiles, E);

  const int mtiles = (M + 15) / 16;
  out_mlp_kernel<<<(mtiles + 7) / 8, 256, 0, stream>>>(
      wsw + 64 * K1P + 64 * 64, b3, b4, out, mtiles, M);
}
